// CosineGatedAttentionUnit_54082228191960
// MI455X (gfx1250) — compile-verified, matrix operand order checked by reading, not run on hardware
//
#include <hip/hip_runtime.h>

// Problem dims (fixed by reference)
#define B_   2
#define N_   2048
#define C_   1024
#define H_   8
#define D_   64
#define DV_  256
#define ROWS (B_*N_)   // 4096
#define C2   (C_*2)    // 2048
#define CE2  (C_*4)    // 4096  (C*E*2)

typedef __attribute__((ext_vector_type(16))) _Float16 v16h;
typedef __attribute__((ext_vector_type(8)))  _Float16 v8h;
typedef __attribute__((ext_vector_type(8)))  float    v8f;

#define NEGF (-3.402823466e38f)

// Build a WMMA f16 operand fragment from two contiguous 16B chunks.
// CDNA5 16-bit A/B layout: lane (l%16) owns the row/col, (l/16) selects the
// K-half; VGPRs 0..3 hold K = kh*8+0..7, VGPRs 4..7 hold K = 16+kh*8+0..7.
__device__ __forceinline__ v16h frag16(const _Float16* lo, const _Float16* hi) {
  v8h a = *(const v8h*)lo;
  v8h b = *(const v8h*)hi;
  v16h r;
#pragma unroll
  for (int i = 0; i < 8; ++i) { r[i] = a[i]; r[i + 8] = b[i]; }
  return r;
}

__device__ __forceinline__ float silu_f(float x) { return x / (1.f + __expf(-x)); }

// Async global->LDS copy of 16 bytes (CDNA5 GLOBAL_LOAD_ASYNC_TO_LDS_B128,
// ASYNCcnt-tracked). LDS byte offset is the low 32 bits of the flat address.
__device__ __forceinline__ void async_cp16(const void* lds_dst, const void* gsrc) {
  uint32_t l = (uint32_t)(uintptr_t)lds_dst;
  uint64_t g = (uint64_t)(uintptr_t)gsrc;
  asm volatile("global_load_async_to_lds_b128 %0, %1, off" :: "v"(l), "v"(g) : "memory");
}
__device__ __forceinline__ void async_wait0() {
  asm volatile("s_wait_asynccnt 0x0" ::: "memory");
}

// ---------------------------------------------------------------------------
// 1) LayerNorm (f32 in) -> f16 activations [ROWS, C]
// ---------------------------------------------------------------------------
__global__ void __launch_bounds__(256)
layernorm_kernel(const float* __restrict__ x, const float* __restrict__ w,
                 const float* __restrict__ b, _Float16* __restrict__ xn) {
  __shared__ float red[256];
  const int row = blockIdx.x;
  const int t   = threadIdx.x;
  const float* xr = x + ((size_t)row << 10);
  float v[4], s = 0.f;
#pragma unroll
  for (int i = 0; i < 4; ++i) { v[i] = xr[t + (i << 8)]; s += v[i]; }
  red[t] = s; __syncthreads();
  for (int o = 128; o > 0; o >>= 1) { if (t < o) red[t] += red[t + o]; __syncthreads(); }
  const float mu = red[0] * (1.f / 1024.f);
  __syncthreads();
  float s2 = 0.f;
#pragma unroll
  for (int i = 0; i < 4; ++i) { const float d = v[i] - mu; s2 += d * d; }
  red[t] = s2; __syncthreads();
  for (int o = 128; o > 0; o >>= 1) { if (t < o) red[t] += red[t + o]; __syncthreads(); }
  const float inv = rsqrtf(red[0] * (1.f / 1024.f) + 1e-5f);
#pragma unroll
  for (int i = 0; i < 4; ++i) {
    const int c = t + (i << 8);
    xn[((size_t)row << 10) + c] = (_Float16)((v[i] - mu) * inv * w[c] + b[c]);
  }
}

// ---------------------------------------------------------------------------
// 2) Tiled weight convert+transpose: W[K][Nc] f32 -> Wt[Nc][K] f16.
//    32x32 LDS tile so both global reads and writes are coalesced (the naive
//    version amplifies HBM traffic ~32x, ~20us at 23.3 TB/s -- not acceptable).
// ---------------------------------------------------------------------------
__global__ void __launch_bounds__(256)
transpose_w_kernel(const float* __restrict__ W, _Float16* __restrict__ Wt,
                   int K, int Nc) {
  __shared__ float tile[32][33];
  const int n0 = blockIdx.x << 5;
  const int k0 = blockIdx.y << 5;
  const int tx = threadIdx.x & 31, ty = threadIdx.x >> 5;  // 32 x 8
#pragma unroll
  for (int i = 0; i < 32; i += 8)
    tile[ty + i][tx] = W[(size_t)(k0 + ty + i) * Nc + n0 + tx];
  __syncthreads();
#pragma unroll
  for (int i = 0; i < 32; i += 8)
    Wt[(size_t)(n0 + ty + i) * K + k0 + tx] = (_Float16)tile[tx][ty + i];
}

// ---------------------------------------------------------------------------
// 3) WMMA GEMM: out[M,Nout] = epilogue(A[M,K] @ Wt[Nout,K]^T + bias)
//    Block tile 64x128, 8 waves, wave tile 16x64 (4 accumulators), K-step 32.
//    Double-buffered LDS filled with GLOBAL_LOAD_ASYNC_TO_LDS_B128 so the
//    next tile's HBM/L2 fetch overlaps the current tile's WMMA stream.
//    MODE 0: silu -> q/k [B,H,N,64]      (p0=q, p1=k)
//    MODE 1: silu -> Vt [B,H,256,N] f16 (p0) / gate f32 [M,2048] (f0)
//    MODE 2: plain -> f32 out [M,Nout]   (f0)
// ---------------------------------------------------------------------------
template <int MODE>
__global__ void __launch_bounds__(256)
gemm16_kernel(const _Float16* __restrict__ A, const _Float16* __restrict__ Wt,
              const float* __restrict__ bias, int K, int Nout,
              _Float16* __restrict__ p0, _Float16* __restrict__ p1,
              float* __restrict__ f0) {
  __shared__ __align__(16) _Float16 As[2][64 * 48];
  __shared__ __align__(16) _Float16 Bs[2][128 * 48];
  const int m0 = blockIdx.x * 64;
  const int n0 = blockIdx.y * 128;
  const int t = threadIdx.x;
  const int wave = t >> 5, lane = t & 31;
  const int wm = (wave & 3) << 4;   // 0,16,32,48
  const int wn = (wave >> 2) << 6;  // 0,64
  const int lr = lane & 15, lh = lane >> 4;

  auto stage = [&](int buf, int k0) {
    {  // A tile: 64 x 32 f16, one b128 per thread
      const int row = t >> 2, kc = (t & 3) << 3;
      async_cp16(&As[buf][row * 48 + kc],
                 A + (size_t)(m0 + row) * K + k0 + kc);
    }
#pragma unroll
    for (int i = 0; i < 2; ++i) {  // B tile: 128 x 32 f16, two b128 per thread
      const int c = (t << 1) + i;
      const int row = c >> 2, kc = (c & 3) << 3;
      async_cp16(&Bs[buf][row * 48 + kc],
                 Wt + (size_t)(n0 + row) * K + k0 + kc);
    }
  };

  v8f acc[4];
#pragma unroll
  for (int i = 0; i < 4; ++i)
#pragma unroll
    for (int r = 0; r < 8; ++r) acc[i][r] = 0.f;

  stage(0, 0);
  for (int k0 = 0; k0 < K; k0 += 32) {
    const int buf = (k0 >> 5) & 1;
    async_wait0();      // only group(buf) outstanding -> drains exactly it
    __syncthreads();
    if (k0 + 32 < K) stage(buf ^ 1, k0 + 32);  // prefetch overlaps compute
    const v16h af = frag16(&As[buf][(wm + lr) * 48 + (lh << 3)],
                           &As[buf][(wm + lr) * 48 + 16 + (lh << 3)]);
#pragma unroll
    for (int i = 0; i < 4; ++i) {
      const int nr = wn + (i << 4) + lr;
      const v16h bf = frag16(&Bs[buf][nr * 48 + (lh << 3)],
                             &Bs[buf][nr * 48 + 16 + (lh << 3)]);
      acc[i] = __builtin_amdgcn_wmma_f32_16x16x32_f16(false, af, false, bf,
                                                      (short)0, acc[i], false, false);
    }
  }

#pragma unroll
  for (int i = 0; i < 4; ++i) {
    const int n = n0 + wn + (i << 4) + lr;
    const float bn = bias[n];
#pragma unroll
    for (int r = 0; r < 8; ++r) {
      const int m = m0 + wm + r + (lh << 3);
      float v = acc[i][r] + bn;
      if (MODE != 2) v = silu_f(v);
      if (MODE == 0) {
        const int b = m >> 11, ns = m & 2047;
        const int h = n >> 7, inner = n & 127;
        _Float16* dst = (inner < 64) ? p0 : p1;
        dst[((((size_t)b * 8 + h) * 2048 + ns) << 6) + (inner & 63)] = (_Float16)v;
      } else if (MODE == 1) {
        const int b = m >> 11, ns = m & 2047;
        if (n < 2048) {
          const int h = n >> 8, dv = n & 255;
          p0[((((size_t)b * 8 + h) << 8) + dv) * 2048 + ns] = (_Float16)v;  // V^T
        } else {
          f0[((size_t)m << 11) + (n - 2048)] = v;  // gate
        }
      } else {
        f0[(size_t)m * Nout + n] = v;
      }
    }
  }
}

// ---------------------------------------------------------------------------
// 4) Per-head l2-norm of q and k rows (D=64), one wave per row (wave32)
// ---------------------------------------------------------------------------
__global__ void __launch_bounds__(256)
l2norm_kernel(_Float16* __restrict__ q, _Float16* __restrict__ k) {
  const int lane = threadIdx.x & 31;
  const int row = blockIdx.x * 8 + (threadIdx.x >> 5);  // 0 .. B*H*N-1
  _Float16* pq = q + ((size_t)row << 6) + (lane << 1);
  _Float16* pk = k + ((size_t)row << 6) + (lane << 1);
  float a0 = (float)pq[0], a1 = (float)pq[1];
  float b0 = (float)pk[0], b1 = (float)pk[1];
  float sq = a0 * a0 + a1 * a1, sk = b0 * b0 + b1 * b1;
#pragma unroll
  for (int off = 1; off < 32; off <<= 1) {
    sq += __shfl_xor(sq, off, 32);
    sk += __shfl_xor(sk, off, 32);
  }
  const float iq = 1.f / fmaxf(sqrtf(sq), 1e-12f);
  const float ik = 1.f / fmaxf(sqrtf(sk), 1e-12f);
  pq[0] = (_Float16)(a0 * iq); pq[1] = (_Float16)(a1 * iq);
  pk[0] = (_Float16)(b0 * ik); pk[1] = (_Float16)(b1 * ik);
}

// ---------------------------------------------------------------------------
// 5) Flash attention (+ gate): block = 128 queries (8 waves x 16), loop over
//    32-key tiles. QK^T: 2 WMMA k-steps (D=64). PV: 16 WMMAs (DV=256, K=32).
//    K/V tiles double-buffered via async-to-LDS; pos_bias (134 MB stream --
//    the HBM floor of this op) prefetched one tile ahead.
// ---------------------------------------------------------------------------
__global__ void __launch_bounds__(256)
attn_kernel(const _Float16* __restrict__ q, const _Float16* __restrict__ k,
            const _Float16* __restrict__ vt, const float* __restrict__ pos_bias,
            const unsigned char* __restrict__ mask, const float* __restrict__ gate,
            const float* __restrict__ temp_p, _Float16* __restrict__ out) {
  const int b = blockIdx.z, h = blockIdx.y;
  const int q0 = blockIdx.x * 128;
  const int t = threadIdx.x, wave = t >> 5, lane = t & 31;
  const int lr = lane & 15, lh = lane >> 4;
  const int qw = q0 + wave * 16;
  const size_t bh = (size_t)b * 8 + h;
  const _Float16* qb = q + bh * 2048 * 64;
  const _Float16* kb = k + bh * 2048 * 64;
  const _Float16* vb = vt + bh * 256 * 2048;
  const float* pb = pos_bias + (size_t)h * 2048 * 2048;
  const unsigned char* mb = mask + ((size_t)b << 11);
  const float temperature = *temp_p;

  __shared__ __align__(16) _Float16 Ks[2][32 * 72];   // 32 keys x 64 (padded)
  __shared__ __align__(16) _Float16 Vs[2][256 * 32];  // 256 dv x 32 keys
  __shared__ __align__(16) _Float16 Ps[8 * 16 * 48];  // per-wave P tiles

  auto stage_kv = [&](int buf, int j0) {
    {  // K tile: 32 x 64 f16, one b128 per thread
      const int row = t >> 2, c = (t & 3) << 3;
      async_cp16(&Ks[buf][row * 72 + c], kb + (size_t)(j0 + row) * 64 + c);
    }
    const _Float16* src = vb + (size_t)t * 2048 + j0;  // V^T: thread = dv row
#pragma unroll
    for (int i = 0; i < 4; ++i)
      async_cp16(&Vs[buf][t * 32 + (i << 3)], src + (i << 3));
  };

  // Query fragments: D=64 -> two K=32 steps
  const _Float16* qrow = qb + (size_t)(qw + lr) * 64;
  const v16h qf0 = frag16(qrow + (lh << 3),      qrow + 16 + (lh << 3));
  const v16h qf1 = frag16(qrow + 32 + (lh << 3), qrow + 48 + (lh << 3));

  int qm[8];
#pragma unroll
  for (int r = 0; r < 8; ++r) qm[r] = mb[qw + r + (lh << 3)];

  float rmax[8], rsum[8];
#pragma unroll
  for (int r = 0; r < 8; ++r) { rmax[r] = NEGF; rsum[r] = 0.f; }
  v8f acc[16];
#pragma unroll
  for (int i = 0; i < 16; ++i)
#pragma unroll
    for (int r = 0; r < 8; ++r) acc[i][r] = 0.f;

  stage_kv(0, 0);
  for (int j0 = 0; j0 < 2048; j0 += 32) {
    const int buf = (j0 >> 5) & 1;
    async_wait0();
    __syncthreads();
    if (j0 + 32 < 2048) {
      stage_kv(buf ^ 1, j0 + 32);
      // prefetch next tile's positional bias rows (emits global_prefetch_b8)
#pragma unroll
      for (int r = 0; r < 8; ++r)
        __builtin_prefetch(&pb[(size_t)(qw + r + (lh << 3)) * 2048 + j0 + 32 + lr], 0, 0);
    }

    // S = q @ k^T for 16 queries x 32 keys (two 16x16 tiles)
    v8f s[2];
#pragma unroll
    for (int tt = 0; tt < 2; ++tt) {
#pragma unroll
      for (int r = 0; r < 8; ++r) s[tt][r] = 0.f;
      const int kr = (tt << 4) + lr;
      const v16h bf0 = frag16(&Ks[buf][kr * 72 + (lh << 3)],
                              &Ks[buf][kr * 72 + 16 + (lh << 3)]);
      s[tt] = __builtin_amdgcn_wmma_f32_16x16x32_f16(false, qf0, false, bf0,
                                                     (short)0, s[tt], false, false);
      const v16h bf1 = frag16(&Ks[buf][kr * 72 + 32 + (lh << 3)],
                              &Ks[buf][kr * 72 + 48 + (lh << 3)]);
      s[tt] = __builtin_amdgcn_wmma_f32_16x16x32_f16(false, qf1, false, bf1,
                                                     (short)0, s[tt], false, false);
    }

    // temperature, positional bias, mask
#pragma unroll
    for (int tt = 0; tt < 2; ++tt) {
      const int col = j0 + (tt << 4) + lr;
      const int km = mb[col];
#pragma unroll
      for (int r = 0; r < 8; ++r) {
        const int row = qw + r + (lh << 3);
        float sv = s[tt][r] * temperature + pb[(size_t)row * 2048 + col];
        if (km | qm[r]) sv = NEGF;
        s[tt][r] = sv;
      }
    }

    // Online softmax: row r lives in vgpr r across the 16 lanes of this half
    float alpha[8];
#pragma unroll
    for (int r = 0; r < 8; ++r) {
      float mx = fmaxf(s[0][r], s[1][r]);
#pragma unroll
      for (int off = 1; off < 16; off <<= 1) mx = fmaxf(mx, __shfl_xor(mx, off, 32));
      const float nm = fmaxf(rmax[r], mx);
      alpha[r] = __expf(rmax[r] - nm);
      rmax[r] = nm;
      const float p0v = __expf(s[0][r] - nm);
      const float p1v = __expf(s[1][r] - nm);
      s[0][r] = p0v; s[1][r] = p1v;
      float sum = p0v + p1v;
#pragma unroll
      for (int off = 1; off < 16; off <<= 1) sum += __shfl_xor(sum, off, 32);
      rsum[r] = rsum[r] * alpha[r] + sum;
    }
#pragma unroll
    for (int i = 0; i < 16; ++i)
#pragma unroll
      for (int r = 0; r < 8; ++r) acc[i][r] *= alpha[r];

    // C/D layout -> A layout via per-wave LDS round trip (wave-internal, so a
    // split-counter DS wait suffices -- no workgroup barrier needed)
    _Float16* pw = &Ps[wave * 16 * 48];
#pragma unroll
    for (int tt = 0; tt < 2; ++tt)
#pragma unroll
      for (int r = 0; r < 8; ++r)
        pw[(r + (lh << 3)) * 48 + (tt << 4) + lr] = (_Float16)s[tt][r];
    asm volatile("s_wait_dscnt 0" ::: "memory");
    const v16h pf = frag16(&pw[lr * 48 + (lh << 3)], &pw[lr * 48 + 16 + (lh << 3)]);

    // acc += P @ V  (K=32 == one WMMA per 16-wide DV tile)
#pragma unroll
    for (int i = 0; i < 16; ++i) {
      const int dvr = (i << 4) + lr;
      const v16h vf = frag16(&Vs[buf][dvr * 32 + (lh << 3)],
                             &Vs[buf][dvr * 32 + 16 + (lh << 3)]);
      acc[i] = __builtin_amdgcn_wmma_f32_16x16x32_f16(false, pf, false, vf,
                                                      (short)0, acc[i], false, false);
    }
  }

  // normalize, gate, store f16 A-matrix for the output GEMM
#pragma unroll
  for (int r = 0; r < 8; ++r) {
    const int row = qw + r + (lh << 3);
    const float inv = 1.f / rsum[r];
#pragma unroll
    for (int i = 0; i < 16; ++i) {
      const int col = h * 256 + (i << 4) + lr;
      const size_t gi = (((size_t)b << 11) + row) * 2048 + col;
      out[gi] = (_Float16)(acc[i][r] * inv * gate[gi]);
    }
  }
}

// ---------------------------------------------------------------------------
// Launch
// ---------------------------------------------------------------------------
extern "C" void kernel_launch(void* const* d_in, const int* in_sizes, int n_in,
                              void* d_out, int out_size, void* d_ws, size_t ws_size,
                              hipStream_t stream) {
  (void)in_sizes; (void)n_in; (void)out_size; (void)ws_size;
  const float* x            = (const float*)d_in[0];
  const unsigned char* mask = (const unsigned char*)d_in[1];  // jnp bool_
  const float* pos_bias     = (const float*)d_in[2];
  const float* ln_w         = (const float*)d_in[3];
  const float* ln_b         = (const float*)d_in[4];
  const float* Wvg          = (const float*)d_in[5];
  const float* bvg          = (const float*)d_in[6];
  const float* Wqk          = (const float*)d_in[7];
  const float* bqk          = (const float*)d_in[8];
  const float* Wo           = (const float*)d_in[9];
  const float* bo           = (const float*)d_in[10];
  const float* temperature  = (const float*)d_in[11];
  float* outp = (float*)d_out;

  char* ws = (char*)d_ws;
  size_t off = 0;
  auto carve = [&](size_t bytes) -> void* {
    void* p = ws + off;
    off += (bytes + 255) & ~(size_t)255;
    return p;
  };
  _Float16* xn   = (_Float16*)carve((size_t)ROWS * C_ * 2);          // 8 MB
  _Float16* WvgT = (_Float16*)carve((size_t)CE2 * C_ * 2);           // 8 MB
  _Float16* WqkT = (_Float16*)carve((size_t)C_ * C_ * 2);            // 2 MB
  _Float16* WoT  = (_Float16*)carve((size_t)C_ * C2 * 2);            // 4 MB
  _Float16* qh   = (_Float16*)carve((size_t)B_ * H_ * N_ * D_ * 2);  // 4 MB
  _Float16* kh   = (_Float16*)carve((size_t)B_ * H_ * N_ * D_ * 2);  // 4 MB
  _Float16* vtb  = (_Float16*)carve((size_t)B_ * H_ * DV_ * N_ * 2); // 16 MB
  float*    gate = (float*)   carve((size_t)ROWS * C2 * 4);          // 32 MB
  _Float16* ag   = (_Float16*)carve((size_t)ROWS * C2 * 2);          // 16 MB

  // 1) LayerNorm -> f16
  layernorm_kernel<<<ROWS, 256, 0, stream>>>(x, ln_w, ln_b, xn);

  // 2) Weight convert+transpose (f32 [K,N] -> f16 [N,K]), tiled/coalesced
  transpose_w_kernel<<<dim3(CE2 / 32, C_ / 32), 256, 0, stream>>>(Wvg, WvgT, C_, CE2);
  transpose_w_kernel<<<dim3(C_ / 32, C_ / 32), 256, 0, stream>>>(Wqk, WqkT, C_, C_);
  transpose_w_kernel<<<dim3(C_ / 32, C2 / 32), 256, 0, stream>>>(Wo, WoT, C2, C_);

  // 3) vg GEMM: [4096,1024]x[1024,4096] -> V^T f16 + gate f32
  gemm16_kernel<1><<<dim3(ROWS / 64, CE2 / 128), 256, 0, stream>>>(
      xn, WvgT, bvg, C_, CE2, vtb, nullptr, gate);

  // 4) qk GEMM: [4096,1024]x[1024,1024] -> q,k f16 [B,H,N,64]
  gemm16_kernel<0><<<dim3(ROWS / 64, C_ / 128), 256, 0, stream>>>(
      xn, WqkT, bqk, C_, C_, qh, kh, nullptr);

  // 5) l2-norm of q/k rows (B*H*N = 32768 rows, 8 waves/block)
  l2norm_kernel<<<(B_ * H_ * N_) / 8, 256, 0, stream>>>(qh, kh);

  // 6) flash attention + gate -> f16 [4096, 2048]
  attn_kernel<<<dim3(N_ / 128, H_, B_), 256, 0, stream>>>(
      qh, kh, vtb, pos_bias, mask, gate, temperature, ag);

  // 7) out GEMM: [4096,2048]x[2048,1024] + bo -> f32 d_out
  gemm16_kernel<2><<<dim3(ROWS / 64, C_ / 128), 256, 0, stream>>>(
      ag, WoT, bo, C2, C_, nullptr, nullptr, outp);
}
